// Head_47906065220078
// MI455X (gfx1250) — matrix operand order repaired and checked by reading, not compiled, not run on hardware
//
#include <hip/hip_runtime.h>
#include <hip/hip_bf16.h>

// Problem constants (reference: B=4, T=2048, C=1024)
#define B_ 4
#define T_ 2048
#define C_ 1024
#define MT_ (B_ * T_)   // 8192 rows of x/q/k/v

typedef __bf16 bf16;
typedef __attribute__((ext_vector_type(4)))  __bf16 v4bf;
typedef __attribute__((ext_vector_type(8)))  __bf16 v8bf;
typedef __attribute__((ext_vector_type(16))) __bf16 v16bf;
typedef __attribute__((ext_vector_type(8)))  float  v8f;

// ---------------------------------------------------------------------------
// Workspace layout (bytes)
//   Xb  : bf16 [MT, C]             16,777,216
//   Wqb : bf16 [C, C]               2,097,152
//   Wkb : bf16 [C, C]               2,097,152
//   Wvb : bf16 [C, C]               2,097,152
//   Qb  : bf16 [MT, C]             16,777,216
//   Kb  : bf16 [MT, C]             16,777,216
//   Vt  : bf16 [C, MT] (V^T!)      16,777,216
//   S   : f32  [B, T, T]           67,108,864
//   P   : bf16 [B, T, T]           33,554,432
// ---------------------------------------------------------------------------
#define OFF_X   ((size_t)0)
#define OFF_WQ  ((size_t)16777216)
#define OFF_WK  (OFF_WQ + (size_t)2097152)
#define OFF_WV  (OFF_WK + (size_t)2097152)
#define OFF_Q   (OFF_WV + (size_t)2097152)
#define OFF_K   (OFF_Q + (size_t)16777216)
#define OFF_VT  (OFF_K + (size_t)16777216)
#define OFF_S   (OFF_VT + (size_t)16777216)
#define OFF_P   (OFF_S + (size_t)67108864)

// Prefetch distance on the K axis: 4 K-panels ahead = 128 elems = 256 bytes
#define PF_ELEMS 128

// ---------------------------------------------------------------------------
// fp32 -> bf16 conversion (4 elements / thread, n always multiple of 4)
// ---------------------------------------------------------------------------
__global__ __launch_bounds__(256)
void cvt_f32_to_bf16(const float* __restrict__ in, bf16* __restrict__ out, int n) {
    int idx = (blockIdx.x * 256 + threadIdx.x) * 4;
    if (idx + 3 < n) {
        float4 f = *(const float4*)(in + idx);
        v4bf o;
        o[0] = (bf16)f.x; o[1] = (bf16)f.y; o[2] = (bf16)f.z; o[3] = (bf16)f.w;
        *(v4bf*)(out + idx) = o;
    }
}

// ---------------------------------------------------------------------------
// A-fragment loader: 16-bit 16x32 A tile, row-major source:
//   lane L<16 : row M=L,    K = {0..15}
//   lane L>=16: row M=L-16, K = {16..31}
// rowptr must already point at row (m0 + (lane&15)).
// ---------------------------------------------------------------------------
__device__ __forceinline__ v16bf load_a_frag(const bf16* rowptr, int k0, int lane) {
    const int kb = (lane >> 4) * 16;
    v16bf a;
    *(v8bf*)&a       = *(const v8bf*)(rowptr + k0 + kb);
    *((v8bf*)&a + 1) = *(const v8bf*)(rowptr + k0 + 8 + kb);
    return a;
}

// ---------------------------------------------------------------------------
// Core register-tiled WMMA GEMM body: wave tile 32(M) x 64(N), 2x4 wmma tiles.
// unroll(4) gives the scheduler a 4-panel window to hoist loads past wmmas;
// WGP-scope prefetches (locality 3) pull the stream into WGP$ ~4 panels ahead.
// ---------------------------------------------------------------------------
#define GEMM_TILE_LOOP(Arow0, Arow1, Brows, k0lo, k0hi, acc)                   \
    _Pragma("unroll 4")                                                        \
    for (int k0 = (k0lo); k0 < (k0hi); k0 += 32) {                             \
        __builtin_prefetch((const void*)((Arow0) + k0 + PF_ELEMS), 0, 3);      \
        __builtin_prefetch((const void*)((Brows)[0] + k0 + PF_ELEMS), 0, 3);   \
        __builtin_prefetch((const void*)((Brows)[2] + k0 + PF_ELEMS), 0, 3);   \
        v16bf a0 = load_a_frag((Arow0), k0, lane);                             \
        v16bf a1 = load_a_frag((Arow1), k0, lane);                             \
        v16bf bf[4];                                                           \
        _Pragma("unroll")                                                      \
        for (int tj = 0; tj < 4; ++tj)                                         \
            bf[tj] = *(const v16bf*)((Brows)[tj] + k0);                        \
        _Pragma("unroll")                                                      \
        for (int tj = 0; tj < 4; ++tj) {                                       \
            acc[0][tj] = __builtin_amdgcn_wmma_f32_16x16x32_bf16(              \
                false, a0, false, bf[tj], (short)0, acc[0][tj], false, false); \
            acc[1][tj] = __builtin_amdgcn_wmma_f32_16x16x32_bf16(              \
                false, a1, false, bf[tj], (short)0, acc[1][tj], false, false); \
        }                                                                      \
    }

// ---------------------------------------------------------------------------
// QKV projection: Y[m, n] = sum_c X[m, c] * W[n, c]   (y = x @ W^T)
// Block = 256 threads = 8 waves in 4(M) x 2(N); block tile 128 x 128.
// outTransposed==0: Y[m*C + n]   (Q, K)
// outTransposed==1: Y[n*MT + m]  (V^T — packed 16B row stores, feeds P@V)
// ---------------------------------------------------------------------------
__global__ __launch_bounds__(256)
void qkv_gemm(const bf16* __restrict__ X, const bf16* __restrict__ W,
              bf16* __restrict__ Y, int outTransposed) {
    const int lane = threadIdx.x & 31;
    const int wave = threadIdx.x >> 5;
    const int m0 = blockIdx.y * 128 + (wave >> 1) * 32;
    const int n0 = blockIdx.x * 128 + (wave & 1) * 64;

    const bf16* Arow0 = X + (size_t)(m0 + (lane & 15)) * C_;
    const bf16* Arow1 = Arow0 + (size_t)16 * C_;
    const bf16* Brows[4];
    #pragma unroll
    for (int tj = 0; tj < 4; ++tj)
        Brows[tj] = W + (size_t)(n0 + tj * 16 + (lane & 15)) * C_ + (lane >> 4) * 16;

    v8f acc[2][4] = {};
    GEMM_TILE_LOOP(Arow0, Arow1, Brows, 0, C_, acc)

    // D layout: lane L -> col N = L&15; VGPR i -> row (L>>4)*8 + i
    #pragma unroll
    for (int ti = 0; ti < 2; ++ti) {
        #pragma unroll
        for (int tj = 0; tj < 4; ++tj) {
            const v8f a = acc[ti][tj];
            const int mb = m0 + ti * 16 + (lane >> 4) * 8;
            const int n  = n0 + tj * 16 + (lane & 15);
            if (outTransposed) {
                v8bf o;
                #pragma unroll
                for (int i = 0; i < 8; ++i) o[i] = (bf16)a[i];
                *(v8bf*)(Y + (size_t)n * MT_ + mb) = o;   // 16B packed store
            } else {
                bf16* Yp = Y + (size_t)mb * C_ + n;
                #pragma unroll
                for (int i = 0; i < 8; ++i) Yp[(size_t)i * C_] = (bf16)a[i];
            }
        }
    }
}

// ---------------------------------------------------------------------------
// Scores: S[b, t, s] = scale * sum_d Q[b,t,d] * K[b,s,d], causally masked.
// Block tile 128(t) x 128(s); wave tile 32x64. Fully-masked blocks/waves skip.
// ---------------------------------------------------------------------------
__global__ __launch_bounds__(256)
void qk_scores(const bf16* __restrict__ Q, const bf16* __restrict__ K,
               float* __restrict__ S) {
    const int b   = blockIdx.z;
    const int t0b = blockIdx.y * 128;
    const int s0b = blockIdx.x * 128;
    if (s0b > t0b + 127) return;           // block fully above the diagonal

    const int lane = threadIdx.x & 31;
    const int wave = threadIdx.x >> 5;
    const int m0 = t0b + (wave >> 1) * 32;   // t
    const int n0 = s0b + (wave & 1) * 64;    // s
    if (n0 > m0 + 31) return;                // wave tile fully masked (uniform)

    const bf16* Arow0 = Q + (size_t)b * T_ * C_ + (size_t)(m0 + (lane & 15)) * C_;
    const bf16* Arow1 = Arow0 + (size_t)16 * C_;
    const bf16* Brows[4];
    #pragma unroll
    for (int tj = 0; tj < 4; ++tj)
        Brows[tj] = K + (size_t)b * T_ * C_
                      + (size_t)(n0 + tj * 16 + (lane & 15)) * C_ + (lane >> 4) * 16;

    v8f acc[2][4] = {};
    GEMM_TILE_LOOP(Arow0, Arow1, Brows, 0, C_, acc)

    const float scale = 0.03125f;  // C^-0.5 = 1/32
    float* Sb = S + (size_t)b * T_ * T_;
    #pragma unroll
    for (int ti = 0; ti < 2; ++ti) {
        #pragma unroll
        for (int tj = 0; tj < 4; ++tj) {
            const v8f a = acc[ti][tj];
            const int mb = m0 + ti * 16 + (lane >> 4) * 8;
            const int s  = n0 + tj * 16 + (lane & 15);
            float* Sp = Sb + (size_t)mb * T_ + s;
            #pragma unroll
            for (int i = 0; i < 8; ++i) {
                const int t = mb + i;
                Sp[(size_t)i * T_] = (s <= t) ? a[i] * scale : -INFINITY;
            }
        }
    }
}

// ---------------------------------------------------------------------------
// Row softmax over S[b, t, 0..t]; writes P bf16, zero-padded to the next
// 64-boundary so P@V can bound its k-loop. One 256-thread block per row.
// ---------------------------------------------------------------------------
__global__ __launch_bounds__(256)
void row_softmax(const float* __restrict__ S, bf16* __restrict__ P) {
    __shared__ float red[8];
    const int row = blockIdx.x;          // b*T + t
    const int b = row >> 11;             // / T_
    const int t = row & (T_ - 1);
    const float* srow = S + (size_t)b * T_ * T_ + (size_t)t * T_;
    bf16*       prow = P + (size_t)b * T_ * T_ + (size_t)t * T_;
    const int n = t + 1;

    float m = -INFINITY;
    for (int i = threadIdx.x; i < n; i += 256) m = fmaxf(m, srow[i]);
    #pragma unroll
    for (int o = 16; o; o >>= 1) m = fmaxf(m, __shfl_xor(m, o, 32));
    if ((threadIdx.x & 31) == 0) red[threadIdx.x >> 5] = m;
    __syncthreads();
    m = red[0];
    #pragma unroll
    for (int i = 1; i < 8; ++i) m = fmaxf(m, red[i]);
    __syncthreads();

    float sum = 0.f;
    for (int i = threadIdx.x; i < n; i += 256) sum += __expf(srow[i] - m);
    #pragma unroll
    for (int o = 16; o; o >>= 1) sum += __shfl_xor(sum, o, 32);
    if ((threadIdx.x & 31) == 0) red[threadIdx.x >> 5] = sum;
    __syncthreads();
    sum = 0.f;
    #pragma unroll
    for (int i = 0; i < 8; ++i) sum += red[i];
    const float inv = 1.0f / sum;

    for (int i = threadIdx.x; i < n; i += 256)
        prow[i] = (bf16)(__expf(srow[i] - m) * inv);
    const int zend = ((t >> 6) + 1) << 6;   // next multiple of 64 (<= T_)
    for (int i = n + threadIdx.x; i < zend; i += 256)
        prow[i] = (bf16)0.f;
}

// ---------------------------------------------------------------------------
// Output: O[b, t, d] = sum_s P[b, t, s] * Vt[d, b*T + s]
// V^T layout makes the B-fragment a contiguous 32B load (same as the other
// GEMMs). k-loop bounded at ceil64(m0+32) per wave (causal range; P rows are
// zero-padded to 64 boundaries by row_softmax).
// ---------------------------------------------------------------------------
__global__ __launch_bounds__(256)
void pv_gemm(const bf16* __restrict__ P, const bf16* __restrict__ Vt,
             float* __restrict__ O) {
    const int b = blockIdx.z;
    const int lane = threadIdx.x & 31;
    const int wave = threadIdx.x >> 5;
    const int m0 = blockIdx.y * 128 + (wave >> 1) * 32;  // t
    const int n0 = blockIdx.x * 128 + (wave & 1) * 64;   // d
    const int kmax = ((m0 + 32 + 63) >> 6) << 6;         // ceil64, <= T_

    const bf16* Arow0 = P + (size_t)b * T_ * T_ + (size_t)(m0 + (lane & 15)) * T_;
    const bf16* Arow1 = Arow0 + (size_t)16 * T_;
    // B row n = d: Vt[d][b*T + k], contiguous in k
    const bf16* Brows[4];
    #pragma unroll
    for (int tj = 0; tj < 4; ++tj)
        Brows[tj] = Vt + (size_t)(n0 + tj * 16 + (lane & 15)) * MT_
                       + (size_t)b * T_ + (lane >> 4) * 16;

    v8f acc[2][4] = {};
    GEMM_TILE_LOOP(Arow0, Arow1, Brows, 0, kmax, acc)

    float* Ob = O + (size_t)b * T_ * C_;
    #pragma unroll
    for (int ti = 0; ti < 2; ++ti) {
        #pragma unroll
        for (int tj = 0; tj < 4; ++tj) {
            const v8f a = acc[ti][tj];
            const int mb = m0 + ti * 16 + (lane >> 4) * 8;
            const int d  = n0 + tj * 16 + (lane & 15);
            float* Op = Ob + (size_t)mb * C_ + d;
            #pragma unroll
            for (int i = 0; i < 8; ++i) Op[(size_t)i * C_] = a[i];
        }
    }
}

// ---------------------------------------------------------------------------
extern "C" void kernel_launch(void* const* d_in, const int* in_sizes, int n_in,
                              void* d_out, int out_size, void* d_ws, size_t ws_size,
                              hipStream_t stream) {
    const float* x  = (const float*)d_in[0];
    const float* Wq = (const float*)d_in[1];
    const float* Wk = (const float*)d_in[2];
    const float* Wv = (const float*)d_in[3];
    float* out = (float*)d_out;

    char* ws = (char*)d_ws;
    bf16* Xb  = (bf16*)(ws + OFF_X);
    bf16* Wqb = (bf16*)(ws + OFF_WQ);
    bf16* Wkb = (bf16*)(ws + OFF_WK);
    bf16* Wvb = (bf16*)(ws + OFF_WV);
    bf16* Qb  = (bf16*)(ws + OFF_Q);
    bf16* Kb  = (bf16*)(ws + OFF_K);
    bf16* Vt  = (bf16*)(ws + OFF_VT);
    float* S  = (float*)(ws + OFF_S);
    bf16* P   = (bf16*)(ws + OFF_P);

    const int nX = MT_ * C_;          // 8,388,608
    const int nW = C_ * C_;           // 1,048,576

    // 1. fp32 -> bf16
    cvt_f32_to_bf16<<<nX / 4 / 256, 256, 0, stream>>>(x, Xb, nX);
    cvt_f32_to_bf16<<<nW / 4 / 256, 256, 0, stream>>>(Wq, Wqb, nW);
    cvt_f32_to_bf16<<<nW / 4 / 256, 256, 0, stream>>>(Wk, Wkb, nW);
    cvt_f32_to_bf16<<<nW / 4 / 256, 256, 0, stream>>>(Wv, Wvb, nW);

    // 2. QKV projections: grid (N/128, M/128) = (8, 64); V stored transposed
    dim3 gProj(C_ / 128, MT_ / 128, 1);
    qkv_gemm<<<gProj, 256, 0, stream>>>(Xb, Wqb, Qb, 0);
    qkv_gemm<<<gProj, 256, 0, stream>>>(Xb, Wkb, Kb, 0);
    qkv_gemm<<<gProj, 256, 0, stream>>>(Xb, Wvb, Vt, 1);

    // 3. S = scale * Q K^T with causal mask: grid (T/128, T/128, B)
    dim3 gScore(T_ / 128, T_ / 128, B_);
    qk_scores<<<gScore, 256, 0, stream>>>(Qb, Kb, S);

    // 4. Row softmax -> P (bf16, zero-padded to 64-boundary)
    row_softmax<<<B_ * T_, 256, 0, stream>>>(S, P);

    // 5. O = P V: grid (C/128, T/128, B)
    dim3 gOut(C_ / 128, T_ / 128, B_);
    pv_gemm<<<gOut, 256, 0, stream>>>(P, Vt, out);
}
